// GINEncoder_59708635349041
// MI455X (gfx1250) — matrix operand order repaired and checked by reading, not compiled, not run on hardware
//
#include <hip/hip_runtime.h>
#include <hip/hip_bf16.h>
#include <stdint.h>

// ---------------------------------------------------------------------------
// GIN encoder for MI455X (gfx1250, wave32).
//  - scatter/pool: wave-per-edge float4 gathers + f32 atomics (L2-resident)
//  - MLP GEMMs:    v_wmma_f32_16x16x32_bf16; W pre-converted once per launch
//                  into bf16 fragment layout, then staged into LDS with
//                  global_load_async_to_lds_b128 (ASYNCcnt path)
//  - BatchNorm:    column reduction (atomics into 1KB stats) + fused bn+relu
// ---------------------------------------------------------------------------

#define N_NODES 100000
#define N_EDGES 1600000
#define N_GRAPHS 512
#define DIM 128
#define NLAYERS 3
#define WFRAG_WORDS 8192  // 4 ktiles * 8 ntiles * 32 lanes * 8 words (32KB)

typedef __attribute__((ext_vector_type(16))) __bf16 v16bf;
typedef __attribute__((ext_vector_type(8)))  float  v8f;

union FragBF {
    v16bf    bf;
    uint32_t u[8];
};

__device__ __forceinline__ uint32_t f2bf_raw(float f) {
    uint32_t u = __float_as_uint(f);
    // round-to-nearest-even bf16 truncation
    uint32_t r = (u + 0x7FFFu + ((u >> 16) & 1u)) >> 16;
    return r & 0xFFFFu;
}
__device__ __forceinline__ uint32_t pack2bf(float lo, float hi) {
    return f2bf_raw(lo) | (f2bf_raw(hi) << 16);
}

// ---------------------------------------------------------------------------
// One-time weight prep: convert all 6 [128x128] fp32 weight matrices into the
// bf16 WMMA-B fragment layout consumed directly by the GEMM's LDS image.
// word index: (((kt*8 + nt)*32 + lane)*8 + v)
//   lane<16 : K = kt*32 + 2v(+1) (+8 if v>=4),     N = nt*16 + lane
//   lane>=16: K = kt*32 + 8 + 2v(+1) (+8 if v>=4), N = nt*16 + (lane-16)
// ---------------------------------------------------------------------------
__global__ __launch_bounds__(256) void gin_prepw(const float* __restrict__ W1,
                                                 const float* __restrict__ W2,
                                                 uint32_t* __restrict__ wfrag) {
    int idx = blockIdx.x * 256 + threadIdx.x;
    if (idx >= 6 * WFRAG_WORDS) return;
    int mat = idx >> 13;          // 0..5 : W1[0..2], W2[0..2]
    int w   = idx & (WFRAG_WORDS - 1);
    int v  = w & 7;
    int l  = (w >> 3) & 31;
    int nt = (w >> 8) & 7;
    int kt = w >> 11;
    int k0  = kt * 32 + ((l & 16) >> 1) + 2 * v + ((v & 4) << 1);
    int col = nt * 16 + (l & 15);
    const float* W = (mat < 3) ? (W1 + (size_t)mat * DIM * DIM)
                               : (W2 + (size_t)(mat - 3) * DIM * DIM);
    wfrag[idx] = pack2bf(W[k0 * DIM + col], W[(k0 + 1) * DIM + col]);
}

// ---------------------------------------------------------------------------
// agg = h (vectorized copy)
// ---------------------------------------------------------------------------
__global__ __launch_bounds__(256) void gin_copy4(const float4* __restrict__ in,
                                                 float4* __restrict__ out, int n4) {
    int i = blockIdx.x * 256 + threadIdx.x;
    if (i < n4) out[i] = in[i];
}

// ---------------------------------------------------------------------------
// agg[dst] += h[src] : one wave per edge, lane covers 4 features
// ---------------------------------------------------------------------------
__global__ __launch_bounds__(256) void gin_scatter(const float* __restrict__ h,
                                                   const int* __restrict__ src,
                                                   const int* __restrict__ dst,
                                                   float* __restrict__ agg, int n_edges) {
    int e = blockIdx.x * 8 + (threadIdx.x >> 5);
    if (e >= n_edges) return;
    int lane = threadIdx.x & 31;
    int s = src[e], d = dst[e];
    const float4 v = *(const float4*)(h + (size_t)s * DIM + lane * 4);
    float* p = agg + (size_t)d * DIM + lane * 4;
    atomicAdd(p + 0, v.x);
    atomicAdd(p + 1, v.y);
    atomicAdd(p + 2, v.z);
    atomicAdd(p + 3, v.w);
}

// ---------------------------------------------------------------------------
// out = [relu](in @ W + bias)   in:[n_rows,128], W pre-laid-out bf16 fragments
// 256 threads = 8 waves; block covers 128 rows; wave computes 16x128 strip.
// 32KB W image staged global->LDS with async-to-LDS (ASYNCcnt), no VGPR pass.
// ---------------------------------------------------------------------------
template <int DO_RELU>
__global__ __launch_bounds__(256) void gin_gemm(const float* __restrict__ in,
                                                const uint32_t* __restrict__ wfrag,
                                                const float* __restrict__ bias,
                                                float* __restrict__ out,
                                                int n_rows) {
    __shared__ uint32_t wlds[WFRAG_WORDS];  // 32KB

    const int tid = threadIdx.x;

    // ---- async stage: 256 threads * 8 issues * 16B = 32KB ---------------
    // LDS byte offset derived from &wlds via ptrtoint: flat LDS-aperture
    // addresses carry the LDS offset in the low 32 bits (ISA aperture map),
    // and the capture makes the "memory"-clobbered asm alias wlds, so the
    // later ds reads cannot be DCE'd.
    {
        uint32_t lds_base = (uint32_t)(uintptr_t)(void*)wlds;
        uint32_t lds_off = lds_base + (uint32_t)tid * 16u;
        uint64_t gaddr = (uint64_t)(uintptr_t)wfrag + (uint64_t)tid * 16u;
#pragma unroll
        for (int it = 0; it < 8; ++it) {
            uint32_t lo = lds_off + (uint32_t)it * 4096u;
            uint64_t ga = gaddr + (uint64_t)it * 4096u;
            asm volatile("global_load_async_to_lds_b128 %0, %1, off"
                         :: "v"(lo), "v"(ga)
                         : "memory");
        }
        asm volatile("s_wait_asynccnt 0x0" ::: "memory");
    }
    __syncthreads();

    const int wave = tid >> 5;
    const int lane = tid & 31;
    const int m0 = blockIdx.x * 128 + wave * 16;

    v8f acc[8];
#pragma unroll
    for (int t = 0; t < 8; ++t) acc[t] = (v8f){0.f, 0.f, 0.f, 0.f, 0.f, 0.f, 0.f, 0.f};

    // A 16x32 (16-bit) layout: lane&15 selects row; lane>=16 -> K offset +8;
    // VGPR v<4: K=2v(+1), v>=4: K=16+2(v-4)(+1)
    int r = m0 + (lane & 15);
    if (r >= n_rows) r = n_rows - 1;  // padded rows compute garbage, never stored
    const float* rowp = in + (size_t)r * DIM;
    const int koff = (lane >= 16) ? 8 : 0;

#pragma unroll
    for (int kt = 0; kt < 4; ++kt) {
        const float* kp = rowp + kt * 32 + koff;
        float4 f0 = *(const float4*)(kp + 0);
        float4 f1 = *(const float4*)(kp + 4);
        float4 f2 = *(const float4*)(kp + 16);
        float4 f3 = *(const float4*)(kp + 20);
        FragBF a;
        a.u[0] = pack2bf(f0.x, f0.y);
        a.u[1] = pack2bf(f0.z, f0.w);
        a.u[2] = pack2bf(f1.x, f1.y);
        a.u[3] = pack2bf(f1.z, f1.w);
        a.u[4] = pack2bf(f2.x, f2.y);
        a.u[5] = pack2bf(f2.z, f2.w);
        a.u[6] = pack2bf(f3.x, f3.y);
        a.u[7] = pack2bf(f3.z, f3.w);

#pragma unroll
        for (int nt = 0; nt < 8; ++nt) {
            const uint32_t* p = &wlds[(((kt * 8 + nt) * 32) + lane) * 8];
            uint4 q0 = *(const uint4*)(p + 0);
            uint4 q1 = *(const uint4*)(p + 4);
            FragBF b;
            b.u[0] = q0.x; b.u[1] = q0.y; b.u[2] = q0.z; b.u[3] = q0.w;
            b.u[4] = q1.x; b.u[5] = q1.y; b.u[6] = q1.z; b.u[7] = q1.w;
            acc[nt] = __builtin_amdgcn_wmma_f32_16x16x32_bf16(
                false, a.bf, false, b.bf, (short)0, acc[nt], false, false);
        }
    }

    // C/D f32 16x16 layout: VGPR i -> row m0 + i + (lane>=16 ? 8 : 0), col = lane&15
    const int rbase = m0 + ((lane >= 16) ? 8 : 0);
    const int coln = lane & 15;
    if (m0 + 16 <= n_rows) {
        // full tile: branch-free coalesced stores
#pragma unroll
        for (int nt = 0; nt < 8; ++nt) {
            const float bv = bias[nt * 16 + coln];
#pragma unroll
            for (int i = 0; i < 8; ++i) {
                float vv = acc[nt][i] + bv;
                if (DO_RELU) vv = fmaxf(vv, 0.f);
                out[(size_t)(rbase + i) * DIM + nt * 16 + coln] = vv;
            }
        }
    } else {
        // ragged last tile
#pragma unroll
        for (int nt = 0; nt < 8; ++nt) {
            const float bv = bias[nt * 16 + coln];
#pragma unroll
            for (int i = 0; i < 8; ++i) {
                int rr = rbase + i;
                if (rr < n_rows) {
                    float vv = acc[nt][i] + bv;
                    if (DO_RELU) vv = fmaxf(vv, 0.f);
                    out[(size_t)rr * DIM + nt * 16 + coln] = vv;
                }
            }
        }
    }
}

// ---------------------------------------------------------------------------
// per-column sum / sum-of-squares (atomics into sums[0..127], sums[128..255])
// ---------------------------------------------------------------------------
__global__ __launch_bounds__(128) void gin_colstats(const float* __restrict__ t,
                                                    int n_rows, int rows_per_block,
                                                    float* __restrict__ sums) {
    int col = threadIdx.x;
    int r0 = blockIdx.x * rows_per_block;
    int r1 = r0 + rows_per_block;
    if (r1 > n_rows) r1 = n_rows;
    float s = 0.f, q = 0.f;
    for (int r = r0; r < r1; ++r) {
        float v = t[(size_t)r * DIM + col];
        s += v;
        q += v * v;
    }
    atomicAdd(&sums[col], s);
    atomicAdd(&sums[DIM + col], q);
}

// ---------------------------------------------------------------------------
// fold mean/var/gamma/beta into per-column scale+shift
// ---------------------------------------------------------------------------
__global__ __launch_bounds__(128) void gin_bnfin(const float* __restrict__ sums,
                                                 const float* __restrict__ gamma,
                                                 const float* __restrict__ beta,
                                                 float* __restrict__ ss, int n_rows) {
    int c = threadIdx.x;
    float inv_n = 1.0f / (float)n_rows;
    float mean = sums[c] * inv_n;
    float var = sums[DIM + c] * inv_n - mean * mean;
    float inv = rsqrtf(var + 1e-5f);
    float sc = inv * gamma[c];
    ss[c] = sc;
    ss[DIM + c] = beta[c] - mean * sc;
}

// ---------------------------------------------------------------------------
// out = relu(t * scale[col] + shift[col]) (vectorized)
// ---------------------------------------------------------------------------
__global__ __launch_bounds__(256) void gin_bnrelu4(const float4* __restrict__ t,
                                                   const float* __restrict__ ss,
                                                   float4* __restrict__ out, int n4) {
    int i = blockIdx.x * 256 + threadIdx.x;
    if (i >= n4) return;
    int c = (i * 4) & (DIM - 1);
    float4 v = t[i];
    float4 r;
    r.x = fmaxf(v.x * ss[c + 0] + ss[DIM + c + 0], 0.f);
    r.y = fmaxf(v.y * ss[c + 1] + ss[DIM + c + 1], 0.f);
    r.z = fmaxf(v.z * ss[c + 2] + ss[DIM + c + 2], 0.f);
    r.w = fmaxf(v.w * ss[c + 3] + ss[DIM + c + 3], 0.f);
    out[i] = r;
}

// ---------------------------------------------------------------------------
// pool[batch[n]] += h[n] : one wave per node
// ---------------------------------------------------------------------------
__global__ __launch_bounds__(256) void gin_pool(const float* __restrict__ h,
                                                const int* __restrict__ batch,
                                                float* __restrict__ pool, int n_nodes) {
    int n = blockIdx.x * 8 + (threadIdx.x >> 5);
    if (n >= n_nodes) return;
    int lane = threadIdx.x & 31;
    int g = batch[n];
    const float4 v = *(const float4*)(h + (size_t)n * DIM + lane * 4);
    float* p = pool + (size_t)g * DIM + lane * 4;
    atomicAdd(p + 0, v.x);
    atomicAdd(p + 1, v.y);
    atomicAdd(p + 2, v.z);
    atomicAdd(p + 3, v.w);
}

// ---------------------------------------------------------------------------

extern "C" void kernel_launch(void* const* d_in, const int* in_sizes, int n_in,
                              void* d_out, int out_size, void* d_ws, size_t ws_size,
                              hipStream_t stream) {
    const size_t ND = (size_t)N_NODES * DIM;

    const float* x     = (const float*)d_in[0];
    const int*   src   = (const int*)d_in[1];
    const int*   dst   = src + N_EDGES;
    const int*   batch = (const int*)d_in[2];
    const float* W1    = (const float*)d_in[3];
    const float* b1    = (const float*)d_in[4];
    const float* W2    = (const float*)d_in[5];
    const float* b2    = (const float*)d_in[6];
    const float* gamma = (const float*)d_in[7];
    const float* beta  = (const float*)d_in[8];

    float*    bufA  = (float*)d_ws;           // agg / t2
    float*    bufB  = bufA + ND;              // mid (after first MLP layer)
    float*    bufC  = bufB + ND;              // h for next layer
    float*    sums  = bufC + ND;              // [256] column sum / sumsq
    float*    ss    = sums + 2 * DIM;         // [256] scale / shift
    uint32_t* wfrag = (uint32_t*)(ss + 2 * DIM);  // 6 * 32KB bf16 fragment images

    float* h_final = (float*)d_out;           // [N_NODES, DIM]
    float* pool    = h_final + ND;            // [N_GRAPHS, DIM]

    const int n4 = (int)(ND / 4);
    const int copy_grid    = (n4 + 255) / 256;
    const int scatter_grid = (N_EDGES + 7) / 8;
    const int gemm_grid    = (N_NODES + 127) / 128;
    const int rows_per_blk = 250;
    const int stats_grid   = (N_NODES + rows_per_blk - 1) / rows_per_blk;

    // one-time bf16 fragment-layout conversion of all 6 weight matrices
    gin_prepw<<<(6 * WFRAG_WORDS + 255) / 256, 256, 0, stream>>>(W1, W2, wfrag);

    const float* hcur = x;
    for (int l = 0; l < NLAYERS; ++l) {
        const float* b1l = b1 + (size_t)l * DIM;
        const float* b2l = b2 + (size_t)l * DIM;
        const float* gl  = gamma + (size_t)l * DIM;
        const float* bl  = beta + (size_t)l * DIM;
        const uint32_t* w1f = wfrag + (size_t)l * WFRAG_WORDS;
        const uint32_t* w2f = wfrag + (size_t)(l + 3) * WFRAG_WORDS;

        // agg = h + segment_sum(h[src] -> dst)
        gin_copy4<<<copy_grid, 256, 0, stream>>>((const float4*)hcur, (float4*)bufA, n4);
        gin_scatter<<<scatter_grid, 256, 0, stream>>>(hcur, src, dst, bufA, N_EDGES);

        // mid = relu(agg @ W1 + b1); t2 = mid @ W2 + b2
        gin_gemm<1><<<gemm_grid, 256, 0, stream>>>(bufA, w1f, b1l, bufB, N_NODES);
        gin_gemm<0><<<gemm_grid, 256, 0, stream>>>(bufB, w2f, b2l, bufA, N_NODES);

        // batchnorm stats over axis 0, then fused bn+relu
        hipMemsetAsync(sums, 0, 2 * DIM * sizeof(float), stream);
        gin_colstats<<<stats_grid, 128, 0, stream>>>(bufA, N_NODES, rows_per_blk, sums);
        gin_bnfin<<<1, 128, 0, stream>>>(sums, gl, bl, ss, N_NODES);

        float* hout = (l == NLAYERS - 1) ? h_final : bufC;
        gin_bnrelu4<<<copy_grid, 256, 0, stream>>>((const float4*)bufA, ss, (float4*)hout, n4);
        hcur = hout;
    }

    // x_pool = segment_sum(h, batch)
    hipMemsetAsync(pool, 0, (size_t)N_GRAPHS * DIM * sizeof(float), stream);
    gin_pool<<<(N_NODES + 7) / 8, 256, 0, stream>>>(h_final, batch, pool, N_NODES);
}
